// GraphNeuralODE_52072183497416
// MI455X (gfx1250) — matrix operand order repaired and checked by reading, not compiled, not run on hardware
//
#include <hip/hip_runtime.h>

typedef _Float16 f16;
typedef __attribute__((ext_vector_type(16))) _Float16 v16h;
typedef __attribute__((ext_vector_type(8)))  _Float16 v8h;
typedef __attribute__((ext_vector_type(8)))  float    v8f;

// Problem constants
#define NB 128   // batch
#define NN 128   // nodes
#define NT 12    // time
#define NF 128   // features
#define NHID 256
#define NH 10

__device__ __forceinline__ v16h cat8(v8h lo, v8h hi) {
  return __builtin_shufflevector(lo, hi, 0,1,2,3,4,5,6,7,8,9,10,11,12,13,14,15);
}

// ---------------------------------------------------------------------------
// Fused GCN layer: Out = act( adj[64-node tile] @ Yfm^T @ W + bias )
//   adjh : f16 [128][128] row-major (row-normalized sigmoid adjacency)
//   yfm  : f16 [b][KIN][128] feature-major activations (B-operand of GEMM1)
//   wt   : f16 [FOUT][KIN] (W transposed; B-operand of GEMM2)
//   obf  : f16 [b][FOUT][128] feature-major output  (when !OUTF32)
//   of32 : f32 [b][FOUT][128] feature-major output  (when OUTF32, no tanh)
// grid = (2 node tiles of 64, 128 batch), block = 256 threads (8 waves)
// Each wave: MT=4 M-tiles x (FOUT/128) N-tiles -> B-fragment reused 4x.
// ---------------------------------------------------------------------------
template<int KIN, int FOUT, bool TANH, bool OUTF32>
__global__ __launch_bounds__(256) void layer_kernel(
    const f16* __restrict__ adjh, const f16* __restrict__ yfm,
    const f16* __restrict__ wt,   const float* __restrict__ bias,
    f16* __restrict__ obf, float* __restrict__ of32)
{
  constexpr int MT = 4;                       // 16-row M tiles per block (64 nodes)
  constexpr int T1 = KIN / 128;               // GEMM1 N-tiles per wave
  constexpr int T2 = FOUT / 128;              // GEMM2 N-tiles per wave
  constexpr int KP = KIN + 8;                 // padded LDS stride

  __shared__ f16  sZ[MT * 16 * KP];           // intermediate Z tile, node-major
  __shared__ float sT[8 * 256];               // per-wave 16x16 transpose staging

  const int b    = blockIdx.y;
  const int m0   = blockIdx.x * (MT * 16);    // node-tile base
  const int lane = threadIdx.x & 31;
  const int wid  = threadIdx.x >> 5;
  const int hl   = lane >> 4;                 // half-wave select
  const int l15  = lane & 15;

  const f16* yb = yfm + (size_t)b * KIN * 128;

  // ---------------- GEMM1: Z[64 x KIN] = adj[m0:+64, :] @ Y[b] ----------------
  v8f c1[T1][MT] = {};
  #pragma unroll
  for (int kk = 0; kk < 128; kk += 32) {
    v16h a[MT];
    #pragma unroll
    for (int mi = 0; mi < MT; mi++) {
      const f16* ap = adjh + (size_t)(m0 + mi * 16 + l15) * 128 + kk + hl * 16;
      a[mi] = cat8(*(const v8h*)ap, *(const v8h*)(ap + 8));
    }
    #pragma unroll
    for (int t = 0; t < T1; t++) {
      const int f = (wid + t * 8) * 16 + l15;               // feature column
      v16h bm = *(const v16h*)(yb + (size_t)f * 128 + kk + hl * 16);
      #pragma unroll
      for (int mi = 0; mi < MT; mi++)
        c1[t][mi] = __builtin_amdgcn_wmma_f32_16x16x32_f16(
            false, a[mi], false, bm, (short)0, c1[t][mi], false, false);
    }
  }
  // spill Z tiles to LDS (node-major f16)
  #pragma unroll
  for (int t = 0; t < T1; t++) {
    const int f = (wid + t * 8) * 16 + l15;
    #pragma unroll
    for (int mi = 0; mi < MT; mi++) {
      #pragma unroll
      for (int r = 0; r < 8; r++) {
        const int m = mi * 16 + r + 8 * hl;
        sZ[m * KP + f] = (f16)c1[t][mi][r];
      }
    }
  }
  __syncthreads();

  // ---------------- GEMM2: Out[64 x FOUT] = Z @ W ----------------
  v8f c2[T2][MT] = {};
  #pragma unroll
  for (int kk = 0; kk < KIN; kk += 32) {
    v16h a[MT];
    #pragma unroll
    for (int mi = 0; mi < MT; mi++) {
      const f16* zp = sZ + (mi * 16 + l15) * KP + kk + hl * 16;
      a[mi] = cat8(*(const v8h*)zp, *(const v8h*)(zp + 8));
    }
    #pragma unroll
    for (int t = 0; t < T2; t++) {
      const int f = (wid + t * 8) * 16 + l15;
      v16h bm = *(const v16h*)(wt + (size_t)f * KIN + kk + hl * 16);
      #pragma unroll
      for (int mi = 0; mi < MT; mi++)
        c2[t][mi] = __builtin_amdgcn_wmma_f32_16x16x32_f16(
            false, a[mi], false, bm, (short)0, c2[t][mi], false, false);
    }
  }

  // ---------------- epilogue: bias + act + transpose to feature-major ----------------
  float* sTw = sT + wid * 256;
  f16*   sTh = (f16*)sTw;
  #pragma unroll
  for (int t = 0; t < T2; t++) {
    const int n0 = (wid + t * 8) * 16;
    const float bv = bias[n0 + l15];
    #pragma unroll
    for (int mi = 0; mi < MT; mi++) {
      #pragma unroll
      for (int r = 0; r < 8; r++) {
        float v = c2[t][mi][r] + bv;
        if constexpr (TANH) v = tanhf(v);
        const int m = r + 8 * hl;
        if constexpr (OUTF32) sTw[l15 * 16 + m] = v;
        else                  sTh[l15 * 16 + m] = (f16)v;
      }
      // same-wave DS ops are in-order: read back transposed, store coalesced
      const int nr = lane >> 1, off = (lane & 1) * 8;
      if constexpr (OUTF32) {
        v8f fv = *(const v8f*)(sTw + nr * 16 + off);
        *(v8f*)(of32 + ((size_t)b * FOUT + n0 + nr) * 128 + m0 + mi * 16 + off) = fv;
      } else {
        v8h hv = *(const v8h*)(sTh + nr * 16 + off);
        *(v8h*)(obf + ((size_t)b * FOUT + n0 + nr) * 128 + m0 + mi * 16 + off) = hv;
      }
    }
  }
}

// ---------------------------------------------------------------------------
// Prep: row-normalized sigmoid adjacency -> f16
// ---------------------------------------------------------------------------
__global__ void prep_adj(const float* __restrict__ aw, f16* __restrict__ adjh) {
  __shared__ float red[128];
  const int n = blockIdx.x, m = threadIdx.x;
  const float s = 1.f / (1.f + expf(-aw[n * 128 + m]));
  red[m] = s;
  __syncthreads();
  for (int st = 64; st > 0; st >>= 1) {
    if (m < st) red[m] += red[m + st];
    __syncthreads();
  }
  const float deg = red[0] < 1.f ? 1.f : red[0];
  adjh[n * 128 + m] = (f16)(s / deg);
}

// W[K][O] (row-major) -> WT f16 [O][K]
template<int K, int O>
__global__ void prep_wt(const float* __restrict__ W, f16* __restrict__ wt) {
  const int i = blockIdx.x * 256 + threadIdx.x;   // i = o*K + k, k fast
  const int o = i / K, k = i % K;
  wt[i] = (f16)W[(size_t)k * O + o];
}

// y0 = x[:,:,-1,:]; write f32 + f16 feature-major state, and out slice h=0
__global__ void init_kernel(const float* __restrict__ x, float* __restrict__ y,
                            f16* __restrict__ ybf, float* __restrict__ out) {
  const int i = blockIdx.x * 256 + threadIdx.x;   // (b, n, f), f fast
  const int f = i & 127, n = (i >> 7) & 127, b = i >> 14;
  const float v = x[((size_t)(b * 128 + n) * NT + (NT - 1)) * 128 + f];
  const size_t yi = ((size_t)(b * 128 + f)) * 128 + n;
  y[yi] = v; ybf[yi] = (f16)v;
  out[((size_t)(b * 128 + n) * NH + 0) * 128 + f] = v;
}

// RK4 combination kernels (flat over feature-major state, 2M elements)
__global__ void combo_a(const float* __restrict__ y, const float* __restrict__ k1,
                        f16* __restrict__ tbf, float c) {
  const int i = blockIdx.x * 256 + threadIdx.x;
  tbf[i] = (f16)(y[i] + c * k1[i]);
}
__global__ void combo_b(const float* __restrict__ y, const float* __restrict__ k1,
                        const float* __restrict__ k2, f16* __restrict__ tbf, float dt) {
  const int i = blockIdx.x * 256 + threadIdx.x;
  tbf[i] = (f16)(y[i] + dt * (k2[i] - k1[i] * (1.f / 3.f)));
}
__global__ void combo_c(const float* __restrict__ y, float* __restrict__ k1,
                        const float* __restrict__ k2, const float* __restrict__ k3,
                        f16* __restrict__ tbf, float dt) {
  const int i = blockIdx.x * 256 + threadIdx.x;
  const float a = k1[i], b2 = k2[i], c3 = k3[i];
  tbf[i] = (f16)(y[i] + dt * (a - b2 + c3));
  k1[i]  = a + 3.f * (b2 + c3);          // fold: k1 + 3*k2 + 3*k3
}
__global__ void final_step(float* __restrict__ y, const float* __restrict__ kacc,
                           const float* __restrict__ k4, f16* __restrict__ ybf,
                           float* __restrict__ out, int h, float dt) {
  const int i = blockIdx.x * 256 + threadIdx.x;
  const float yn = y[i] + dt * (kacc[i] + k4[i]) * 0.125f;
  y[i] = yn; ybf[i] = (f16)yn;
  const int n = i & 127, f = (i >> 7) & 127, b = i >> 14;
  out[((size_t)(b * 128 + n) * NH + h) * 128 + f] = yn;
}

// ---------------------------------------------------------------------------
extern "C" void kernel_launch(void* const* d_in, const int* in_sizes, int n_in,
                              void* d_out, int out_size, void* d_ws, size_t ws_size,
                              hipStream_t stream) {
  const float* x  = (const float*)d_in[0];
  const float* aw = (const float*)d_in[1];
  const float* W1 = (const float*)d_in[2];
  const float* b1 = (const float*)d_in[3];
  const float* W2 = (const float*)d_in[4];
  const float* b2 = (const float*)d_in[5];
  const float* W3 = (const float*)d_in[6];
  const float* b3 = (const float*)d_in[7];
  float* out = (float*)d_out;

  const size_t SB = (size_t)NB * NF * NN;      // 2,097,152 f32 state elems
  const size_t SH = (size_t)NB * NHID * NN;    // hidden f16 elems

  char* w = (char*)d_ws;
  float* y  = (float*)w;  w += SB * 4;
  float* k1 = (float*)w;  w += SB * 4;
  float* k2 = (float*)w;  w += SB * 4;
  float* kc = (float*)w;  w += SB * 4;
  f16* ybf  = (f16*)w;    w += SB * 2;
  f16* tbf  = (f16*)w;    w += SB * 2;
  f16* h1   = (f16*)w;    w += SH * 2;
  f16* h2   = (f16*)w;    w += SH * 2;
  f16* adjh = (f16*)w;    w += (size_t)NN * NN * 2;
  f16* wt1  = (f16*)w;    w += (size_t)NHID * NF * 2;   // [256][128]
  f16* wt2  = (f16*)w;    w += (size_t)NHID * NHID * 2; // [256][256]
  f16* wt3  = (f16*)w;    w += (size_t)NF * NHID * 2;   // [128][256]

  const float dt = 10.0f / 9.0f;
  const dim3 lg(2, 128);                       // 64-node tiles x batch
  const int EW = (int)(SB / 256);

  prep_adj<<<128, 128, 0, stream>>>(aw, adjh);
  prep_wt<128, 256><<<(128 * 256) / 256, 256, 0, stream>>>(W1, wt1);
  prep_wt<256, 256><<<(256 * 256) / 256, 256, 0, stream>>>(W2, wt2);
  prep_wt<256, 128><<<(256 * 128) / 256, 256, 0, stream>>>(W3, wt3);
  init_kernel<<<EW, 256, 0, stream>>>(x, y, ybf, out);

  auto run_f = [&](const f16* in, float* kout) {
    layer_kernel<128, 256, true,  false><<<lg, 256, 0, stream>>>(adjh, in, wt1, b1, h1, nullptr);
    layer_kernel<256, 256, true,  false><<<lg, 256, 0, stream>>>(adjh, h1, wt2, b2, h2, nullptr);
    layer_kernel<256, 128, false, true ><<<lg, 256, 0, stream>>>(adjh, h2, wt3, b3, nullptr, kout);
  };

  for (int h = 1; h < NH; h++) {
    run_f(ybf, k1);                                            // k1 = f(y)
    combo_a<<<EW, 256, 0, stream>>>(y, k1, tbf, dt / 3.0f);    // y + dt*k1/3
    run_f(tbf, k2);                                            // k2
    combo_b<<<EW, 256, 0, stream>>>(y, k1, k2, tbf, dt);       // y + dt*(k2 - k1/3)
    run_f(tbf, kc);                                            // k3
    combo_c<<<EW, 256, 0, stream>>>(y, k1, k2, kc, tbf, dt);   // y + dt*(k1-k2+k3); k1+=3(k2+k3)
    run_f(tbf, kc);                                            // k4
    final_step<<<EW, 256, 0, stream>>>(y, k1, kc, ybf, out, h, dt);
  }
}
